// Attention_21732534518476
// MI455X (gfx1250) — matrix operand order repaired and checked by reading, not compiled, not run on hardware
//
#include <hip/hip_runtime.h>

typedef __attribute__((ext_vector_type(16))) __bf16 v16bf;
typedef __attribute__((ext_vector_type(8)))  float  v8f;
typedef __attribute__((ext_vector_type(8)))  unsigned short ushort8;

static constexpr int kB    = 4;
static constexpr int kN    = 2048;   // sequence length
static constexpr int kC    = 256;    // channels
static constexpr int kH    = 8;      // heads
static constexpr int kD    = 32;     // dim per head
static constexpr int kQKV  = 768;    // 3 * H * D
static constexpr int kM    = kB * kN; // 8192 rows

__device__ __forceinline__ unsigned short f2bf(float f) {
    unsigned u = __float_as_uint(f);
    u += 0x7FFFu + ((u >> 16) & 1u);          // round-to-nearest-even
    return (unsigned short)(u >> 16);
}

union Frag { v16bf v; ushort8 h[2]; unsigned short u[16]; };

// A fragment: 16x32 tile, row-major with given stride (elements).
// Lane l: row = l&15, half = l>>4. VGPR0-3: K = half*8 .. +7, VGPR4-7: K = 16+half*8 .. +7
__device__ __forceinline__ v16bf load_a_frag(const unsigned short* base, int stride, int lane) {
    int row = lane & 15, half = lane >> 4;
    const unsigned short* p = base + row * stride;
    Frag f;
    f.h[0] = *reinterpret_cast<const ushort8*>(p + half * 8);
    f.h[1] = *reinterpret_cast<const ushort8*>(p + 16 + half * 8);
    return f.v;
}

// B fragment: 32x16 tile taken from an N-major (n x k) buffer with given stride.
// Lane l: col n = l&15, khalf = l>>4. VGPR0-3: K = khalf*8 .. +7, VGPR4-7: K = 16+khalf*8 .. +7
__device__ __forceinline__ v16bf load_b_frag(const unsigned short* base, int stride, int lane) {
    int n = lane & 15, khalf = lane >> 4;
    const unsigned short* p = base + n * stride + khalf * 8;
    Frag f;
    f.h[0] = *reinterpret_cast<const ushort8*>(p);
    f.h[1] = *reinterpret_cast<const ushort8*>(p + 16);
    return f.v;
}

__device__ __forceinline__ void wmma_acc(v8f& c, v16bf a, v16bf b) {
    c = __builtin_amdgcn_wmma_f32_16x16x32_bf16(false, a, false, b, (short)0, c, false, false);
}

// ---------------- elementwise fp32 -> bf16 ----------------
__global__ __launch_bounds__(256) void cvt_bf16_kernel(const float* __restrict__ src,
                                                       unsigned short* __restrict__ dst, int n) {
    int i = blockIdx.x * blockDim.x + threadIdx.x;
    if (i < n) dst[i] = f2bf(src[i]);
}

// transpose [K x Nc] fp32 -> [Nc x K] bf16   (K = 256)
__global__ __launch_bounds__(256) void transpose_bf16_kernel(const float* __restrict__ src,
                                                             unsigned short* __restrict__ dst,
                                                             int K, int Nc) {
    int tid = blockIdx.x * blockDim.x + threadIdx.x;
    if (tid >= K * Nc) return;
    int n = tid / K, k = tid % K;
    dst[tid] = f2bf(src[k * Nc + n]);
}

// ---------------- QKV projection GEMM (bf16 WMMA, fp32 acc) ----------------
// xb:[8192][256] bf16, wT:[768][256] bf16.
// Scatters into qs (scaled, key-major), ks (key-major), vT (d-major) bf16.
__global__ __launch_bounds__(128) void qkv_gemm_kernel(const unsigned short* __restrict__ xb,
                                                       const unsigned short* __restrict__ wT,
                                                       unsigned short* __restrict__ qs,
                                                       unsigned short* __restrict__ ks,
                                                       unsigned short* __restrict__ vT) {
    int lane = threadIdx.x & 31, wave = threadIdx.x >> 5;
    int m0 = blockIdx.x * 16;
    int n_base = (blockIdx.y * 4 + wave) * 64;

    v8f c[4] = {};
    for (int k0 = 0; k0 < kC; k0 += 32) {
        v16bf a = load_a_frag(xb + m0 * kC + k0, kC, lane);
        #pragma unroll
        for (int t = 0; t < 4; ++t) {
            v16bf b = load_b_frag(wT + (n_base + 16 * t) * kC + k0, kC, lane);
            wmma_acc(c[t], a, b);
        }
    }
    const float scale = 0.17677669529663687f; // 32^-0.5
    int col = lane & 15, half = lane >> 4;
    #pragma unroll
    for (int t = 0; t < 4; ++t) {
        int j = n_base + 16 * t + col;
        #pragma unroll
        for (int r = 0; r < 8; ++r) {
            int m  = m0 + r + 8 * half;
            int b  = m >> 11;           // batch
            int nn = m & (kN - 1);      // sequence position
            float val = c[t][r];
            if (j < 256) {
                int h = j >> 5, d = j & 31;
                qs[((b * kH + h) * kN + nn) * kD + d] = f2bf(val * scale);
            } else if (j < 512) {
                int jj = j - 256; int h = jj >> 5, d = jj & 31;
                ks[((b * kH + h) * kN + nn) * kD + d] = f2bf(val);
            } else {
                int jj = j - 512; int h = jj >> 5, d = jj & 31;
                vT[((b * kH + h) * kD + d) * kN + nn] = f2bf(val);
            }
        }
    }
}

// ---------------- fused flash attention (transposed-score scheme) ----------------
// One wave handles a 16-query block of one (b,h); loops over 32-key tiles.
// S^T = K * Q^T so that softmax'd probabilities land directly in the A-fragment
// layout needed for P*V — no LDS transpose required.
__global__ __launch_bounds__(128) void attn_kernel(const unsigned short* __restrict__ qs,
                                                   const unsigned short* __restrict__ ks,
                                                   const unsigned short* __restrict__ vT,
                                                   const float* __restrict__ bias_table,
                                                   const int* __restrict__ rel_index,
                                                   unsigned short* __restrict__ ob) {
    int lane = threadIdx.x & 31, wave = threadIdx.x >> 5;
    int bh = blockIdx.x >> 5;                    // (b*H + h)
    int qb = ((blockIdx.x & 31) << 2) + wave;    // 16-row query block index
    int b  = bh >> 3, h = bh & 7;
    int col = lane & 15, half = lane >> 4;

    const unsigned short* kbh = ks + bh * kN * kD;   // [key][d]
    const unsigned short* vbh = vT + bh * kD * kN;   // [d][key]

    // Q^T as B fragment: lane = query column, contiguous d
    v16bf qtf = load_b_frag(qs + (bh * kN + qb * 16) * kD, kD, lane);

    v8f acc0 = {}, acc1 = {};                    // out: lane = d col, VGPR = query row
    float m_run = -3.0e38f, l_run = 0.0f;        // stats for query (qb*16 + col)

    int q_global = qb * 16 + col;
    const int* relrow = rel_index + q_global * kN;

    for (int kb = 0; kb < kN; kb += 32) {
        __builtin_prefetch(kbh + (kb + 32) * kD, 0, 0);
        __builtin_prefetch(vbh + kb + 32, 0, 0);

        // S^T tiles: A = K tile (16 keys x 32 d), B = Q^T
        v16bf ka0 = load_a_frag(kbh + (kb + 0)  * kD, kD, lane);
        v16bf ka1 = load_a_frag(kbh + (kb + 16) * kD, kD, lane);
        v8f s0 = {}, s1 = {};
        wmma_acc(s0, ka0, qtf);   // keys kb+8*half+{0..7}      for query col
        wmma_acc(s1, ka1, qtf);   // keys kb+16+8*half+{0..7}   for query col

        // relative-position bias: keys are consecutive per lane -> int4 loads
        const int* rp = relrow + kb + 8 * half;
        int4 ia = *reinterpret_cast<const int4*>(rp);
        int4 ib = *reinterpret_cast<const int4*>(rp + 4);
        int4 ic = *reinterpret_cast<const int4*>(rp + 16);
        int4 id = *reinterpret_cast<const int4*>(rp + 20);
        int idx0[8] = {ia.x, ia.y, ia.z, ia.w, ib.x, ib.y, ib.z, ib.w};
        int idx1[8] = {ic.x, ic.y, ic.z, ic.w, id.x, id.y, id.z, id.w};
        #pragma unroll
        for (int r = 0; r < 8; ++r) {
            s0[r] += bias_table[idx0[r] * kH + h];
            s1[r] += bias_table[idx1[r] * kH + h];
        }

        // online softmax: row = query lives in this lane (+ its half-partner)
        float mx = fmaxf(s0[0], s1[0]);
        #pragma unroll
        for (int r = 1; r < 8; ++r) mx = fmaxf(mx, fmaxf(s0[r], s1[r]));
        mx = fmaxf(mx, __shfl_xor(mx, 16, 32));
        float mn   = fmaxf(m_run, mx);
        float corr = __expf(m_run - mn);
        float sum  = 0.0f;
        Frag pf;
        #pragma unroll
        for (int r = 0; r < 8; ++r) {
            float e0 = __expf(s0[r] - mn);
            float e1 = __expf(s1[r] - mn);
            sum += e0 + e1;
            pf.u[r]     = f2bf(e0);   // A-frag elems 0..7  : keys 8*half+{0..7}
            pf.u[8 + r] = f2bf(e1);   // A-frag elems 8..15 : keys 16+8*half+{0..7}
        }
        sum += __shfl_xor(sum, 16, 32);
        l_run = l_run * corr + sum;
        m_run = mn;

        // rescale output accumulators: broadcast per-query factor to all lanes
        #pragma unroll
        for (int r = 0; r < 8; ++r) {
            float cr = __shfl(corr, r + 8 * half, 32);
            acc0[r] *= cr;
            acc1[r] *= cr;
        }

        v16bf vf0 = load_b_frag(vbh + kb, kN, lane);            // d cols 0..15
        v16bf vf1 = load_b_frag(vbh + 16 * kN + kb, kN, lane);  // d cols 16..31
        wmma_acc(acc0, pf.v, vf0);
        wmma_acc(acc1, pf.v, vf1);
    }

    // normalize and store bf16 output, laid out [b][n][h*32 + d]
    float linv = 1.0f / l_run;
    #pragma unroll
    for (int r = 0; r < 8; ++r) {
        float inv = __shfl(linv, r + 8 * half, 32);
        int nn = qb * 16 + r + 8 * half;
        unsigned short* orow = ob + (b * kN + nn) * kC + h * kD;
        orow[col]      = f2bf(acc0[r] * inv);
        orow[16 + col] = f2bf(acc1[r] * inv);
    }
}

// ---------------- output projection GEMM + bias ----------------
__global__ __launch_bounds__(128) void out_gemm_kernel(const unsigned short* __restrict__ ob,
                                                       const unsigned short* __restrict__ wT,
                                                       const float* __restrict__ bias,
                                                       float* __restrict__ out) {
    int lane = threadIdx.x & 31, wave = threadIdx.x >> 5;
    int m0 = blockIdx.x * 16;
    int n_base = wave * 64;

    v8f c[4] = {};
    for (int k0 = 0; k0 < kC; k0 += 32) {
        v16bf a = load_a_frag(ob + m0 * kC + k0, kC, lane);
        #pragma unroll
        for (int t = 0; t < 4; ++t) {
            v16bf b = load_b_frag(wT + (n_base + 16 * t) * kC + k0, kC, lane);
            wmma_acc(c[t], a, b);
        }
    }
    int col = lane & 15, half = lane >> 4;
    #pragma unroll
    for (int t = 0; t < 4; ++t) {
        int j = n_base + 16 * t + col;
        float bj = bias[j];
        #pragma unroll
        for (int r = 0; r < 8; ++r) {
            int m = m0 + r + 8 * half;
            out[m * kC + j] = c[t][r] + bj;
        }
    }
}

extern "C" void kernel_launch(void* const* d_in, const int* in_sizes, int n_in,
                              void* d_out, int out_size, void* d_ws, size_t ws_size,
                              hipStream_t stream) {
    const float* x          = (const float*)d_in[0];
    const float* w_qkv      = (const float*)d_in[1];
    const float* bias_table = (const float*)d_in[2];
    const float* w_out      = (const float*)d_in[3];
    const float* b_out      = (const float*)d_in[4];
    const int*   rel_index  = (const int*)d_in[5];
    float* out = (float*)d_out;

    char* ws = (char*)d_ws;
    size_t off = 0;
    auto alloc = [&](size_t bytes) -> void* {
        void* p = ws + off;
        off += (bytes + 255) & ~(size_t)255;
        return p;
    };
    unsigned short* xb     = (unsigned short*)alloc((size_t)kM * kC * 2);
    unsigned short* wqkvT  = (unsigned short*)alloc((size_t)kQKV * kC * 2);
    unsigned short* woutT  = (unsigned short*)alloc((size_t)kC * kC * 2);
    unsigned short* qsb    = (unsigned short*)alloc((size_t)kB * kH * kN * kD * 2);
    unsigned short* ksb    = (unsigned short*)alloc((size_t)kB * kH * kN * kD * 2);
    unsigned short* vTb    = (unsigned short*)alloc((size_t)kB * kH * kN * kD * 2);
    unsigned short* obuf   = (unsigned short*)alloc((size_t)kM * kC * 2);

    int nx = kM * kC;
    cvt_bf16_kernel<<<(nx + 255) / 256, 256, 0, stream>>>(x, xb, nx);
    transpose_bf16_kernel<<<(kC * kQKV + 255) / 256, 256, 0, stream>>>(w_qkv, wqkvT, kC, kQKV);
    transpose_bf16_kernel<<<(kC * kC + 255) / 256, 256, 0, stream>>>(w_out, woutT, kC, kC);

    qkv_gemm_kernel<<<dim3(kM / 16, kQKV / 256), 128, 0, stream>>>(xb, wqkvT, qsb, ksb, vTb);

    attn_kernel<<<kB * kH * (kN / 64), 128, 0, stream>>>(qsb, ksb, vTb, bias_table, rel_index, obuf);

    out_gemm_kernel<<<dim3(kM / 16, 1), 128, 0, stream>>>(obuf, woutT, b_out, out);
}
